// BitConv1d_35605278884138
// MI455X (gfx1250) — 8-bit matrix operand order repaired and checked by reading, compiled, not run on hardware
//
#include <hip/hip_runtime.h>
#include <stdint.h>

typedef int v8i __attribute__((ext_vector_type(8)));

#define USE_ASYNC_STAGE 1

#define QPF 127.0f
#define EPS_NORM  1e-6f
#define EPS_SCALE 1e-5f

#define B_   8
#define CIN  512
#define T_   8192
#define COUT 512
#define KW   7

#define WS_SCALEBITS 0u
#define WS_WSCALE    16u
#define WS_FINAL     20u
#define WS_PARTIAL   64u
#define WS_INVRMS    4096u
#define WS_XQ        524288u
#define WS_WPACK     (WS_XQ + (unsigned)(B_*CIN*T_))

__global__ void k_init(unsigned* scalebits) {
    if (threadIdx.x == 0) *scalebits = 0u;
}

__global__ void k_rms_max(const float* __restrict__ x, const float* __restrict__ gamma,
                          float* __restrict__ inv_rms, unsigned* __restrict__ scalebits) {
    int gid = blockIdx.x * 256 + threadIdx.x;
    int b = gid >> 13;
    int t = gid & (T_ - 1);
    const float* xp = x + ((size_t)b * CIN) * T_ + t;
    float ss = 0.f, amax = 0.f;
    for (int c = 0; c < CIN; ++c) {
        float v = xp[(size_t)c * T_];
        float g = gamma[c];
        ss = fmaf(v, v, ss);
        amax = fmaxf(amax, fabsf(v * g));
    }
    float ir = 1.0f / sqrtf(ss * (1.0f / CIN) + EPS_NORM);
    inv_rms[gid] = ir;
    float sc = amax * ir;
    __shared__ float red[256];
    red[threadIdx.x] = sc;
    __syncthreads();
    for (int s = 128; s > 0; s >>= 1) {
        if ((int)threadIdx.x < s) red[threadIdx.x] = fmaxf(red[threadIdx.x], red[threadIdx.x + s]);
        __syncthreads();
    }
    if (threadIdx.x == 0) atomicMax(scalebits, __float_as_uint(red[0]));
}

__global__ void k_wsum(const float* __restrict__ w, float* __restrict__ partial, int n) {
    float s = 0.f;
    for (int i = blockIdx.x * 256 + threadIdx.x; i < n; i += 65536) s += fabsf(w[i]);
    __shared__ float red[256];
    red[threadIdx.x] = s;
    __syncthreads();
    for (int k = 128; k > 0; k >>= 1) {
        if ((int)threadIdx.x < k) red[threadIdx.x] += red[threadIdx.x + k];
        __syncthreads();
    }
    if (threadIdx.x == 0) partial[blockIdx.x] = red[0];
}

__global__ void k_finalize(const float* __restrict__ partial, const unsigned* __restrict__ scalebits,
                           float* __restrict__ wscale, float* __restrict__ finalscale, int n) {
    __shared__ float red[256];
    red[threadIdx.x] = partial[threadIdx.x];
    __syncthreads();
    for (int k = 128; k > 0; k >>= 1) {
        if ((int)threadIdx.x < k) red[threadIdx.x] += red[threadIdx.x + k];
        __syncthreads();
    }
    if (threadIdx.x == 0) {
        float mean = red[0] / (float)n;
        float wsv = fmaxf(mean, EPS_SCALE);
        *wscale = wsv;
        float asv = fmaxf(__uint_as_float(*scalebits), EPS_SCALE);
        *finalscale = (asv / QPF) * wsv;
    }
}

__device__ __forceinline__ unsigned quant8(float v) {
    v = fminf(fmaxf(v, -128.0f), 127.0f);
    return (unsigned)((int)rintf(v) & 0xFF);
}

__global__ __launch_bounds__(256) void k_actq(const float* __restrict__ x,
                                              const float* __restrict__ gamma,
                                              const float* __restrict__ inv_rms,
                                              const unsigned* __restrict__ scalebits,
                                              signed char* __restrict__ xq_t) {
    __shared__ unsigned lq[64 * 16];
    unsigned bid = blockIdx.x;
    unsigned cig = bid & 7u;
    unsigned tg  = (bid >> 3) & 127u;
    unsigned b   = bid >> 10;
    unsigned ci0 = cig * 64u;
    unsigned t0  = tg * 64u;
    unsigned tid = threadIdx.x;

    float s = fmaxf(__uint_as_float(*scalebits), EPS_SCALE);
    float qs = QPF / s;

    unsigned ci = tid >> 2;
    float g = gamma[ci0 + ci] * qs;
    const float* xp = x + ((size_t)(b * CIN + ci0 + ci)) * T_ + t0;
    const float* ip = inv_rms + ((size_t)b << 13) + t0;
    #pragma unroll
    for (int it = 0; it < 4; ++it) {
        unsigned chunk = (tid & 3u) + (unsigned)it * 4u;
        float4 xv = *(const float4*)(xp + chunk * 4u);
        float4 iv = *(const float4*)(ip + chunk * 4u);
        unsigned p = quant8(xv.x * iv.x * g)
                   | (quant8(xv.y * iv.y * g) << 8)
                   | (quant8(xv.z * iv.z * g) << 16)
                   | (quant8(xv.w * iv.w * g) << 24);
        lq[ci * 16u + chunk] = p;
    }
    __syncthreads();
    const unsigned char* lb = (const unsigned char*)lq;
    #pragma unroll
    for (int it = 0; it < 4; ++it) {
        unsigned idx = tid + (unsigned)it * 256u;
        unsigned t = idx >> 4;
        unsigned cq = idx & 15u;
        unsigned p = (unsigned)lb[(cq * 4u + 0) * 64u + t]
                   | ((unsigned)lb[(cq * 4u + 1) * 64u + t] << 8)
                   | ((unsigned)lb[(cq * 4u + 2) * 64u + t] << 16)
                   | ((unsigned)lb[(cq * 4u + 3) * 64u + t] << 24);
        *(unsigned*)(xq_t + ((size_t)(b * T_ + t0 + t)) * CIN + ci0 + cq * 4u) = p;
    }
}

__global__ void k_wpack(const float* __restrict__ w, const float* __restrict__ wscale,
                        unsigned* __restrict__ wpack) {
    unsigned d = blockIdx.x * 256 + threadIdx.x;
    if (d >= 7u * 65536u) return;
    unsigned v    = d & 7u;
    unsigned lane = (d >> 3) & 31u;
    unsigned cic  = (d >> 8) & 7u;
    unsigned cot  = (d >> 11) & 31u;
    unsigned k    = d >> 16;
    unsigned co = cot * 16u + (lane & 15u);
    unsigned hi = lane >> 4;
    unsigned ci = cic * 64u + ((v >> 2) << 5) + hi * 16u + (v & 3u) * 4u;
    float inv = 1.0f / *wscale;
    unsigned pack = 0u;
    for (int j = 0; j < 4; ++j) {
        float wn = w[((size_t)co * CIN + ci + j) * KW + k] * inv;
        wn = fminf(fmaxf(wn, -1.0f), 1.0f);
        int q = (int)rintf(wn);
        pack |= ((unsigned)(q & 0xFF)) << (8 * j);
    }
    wpack[d] = pack;
}

#define LSTRIDE 528u
__global__ __launch_bounds__(256) void k_conv(const signed char* __restrict__ xq_t,
                                              const unsigned* __restrict__ wpack,
                                              const float* __restrict__ finalscale,
                                              float* __restrict__ out) {
    __shared__ __align__(16) unsigned char xs[136 * LSTRIDE];
    unsigned bid = blockIdx.x;
    unsigned mt = bid & 3u;
    unsigned tt = (bid >> 2) & 63u;
    unsigned b  = bid >> 8;
    int t0 = (int)(tt * 128u);
    unsigned tid = threadIdx.x;
    unsigned lane = tid & 31u, wv = tid >> 5;
    unsigned cot_g = mt * 8u + wv;
    unsigned hi16 = (lane >> 4) * 16u;

    const signed char* gbase = xq_t + (size_t)b * T_ * CIN;

    #pragma unroll
    for (int it = 0; it < 17; ++it) {
        unsigned tau = tid + (unsigned)it * 256u;
        unsigned chunk = tau & 31u;
        unsigned row = tau >> 5;
        int t = t0 - 4 + (int)row;
        unsigned ldsoff = row * LSTRIDE + chunk * 16u;
        if (t >= 0 && t < T_) {
#if USE_ASYNC_STAGE
            unsigned lv = (unsigned)(uintptr_t)&xs[ldsoff];
            unsigned voff = (unsigned)t * (unsigned)CIN + chunk * 16u;
            asm volatile("global_load_async_to_lds_b128 %0, %1, %2"
                         :: "v"(lv), "v"(voff), "s"(gbase) : "memory");
#else
            *(uint4*)(&xs[ldsoff]) = *(const uint4*)(gbase + (size_t)t * CIN + chunk * 16u);
#endif
        } else {
            *(uint4*)(&xs[ldsoff]) = make_uint4(0u, 0u, 0u, 0u);
        }
    }
#if USE_ASYNC_STAGE
    asm volatile("s_wait_asynccnt 0" ::: "memory");
#endif
    __syncthreads();

    v8i acc[8];
    #pragma unroll
    for (int i = 0; i < 8; ++i) acc[i] = (v8i){0,0,0,0,0,0,0,0};

    for (int cic = 0; cic < 8; ++cic) {
        #pragma unroll
        for (int k = 0; k < 7; ++k) {
            const uint4* ap = (const uint4*)(wpack +
                ((((size_t)k * 32 + cot_g) * 8 + (unsigned)cic) * 256) + lane * 8);
            uint4 a0 = ap[0], a1 = ap[1];
            v8i av;
            av[0] = (int)a0.x; av[1] = (int)a0.y; av[2] = (int)a0.z; av[3] = (int)a0.w;
            av[4] = (int)a1.x; av[5] = (int)a1.y; av[6] = (int)a1.z; av[7] = (int)a1.w;
            unsigned cbase = (unsigned)cic * 64u + hi16;
            #pragma unroll
            for (int ni = 0; ni < 8; ++ni) {
                unsigned row = (unsigned)(ni * 16) + (lane & 15u) + (unsigned)k + 1u;
                uint4 b0 = *(const uint4*)(&xs[row * LSTRIDE + cbase]);
                uint4 b1 = *(const uint4*)(&xs[row * LSTRIDE + cbase + 32u]);
                v8i bv;
                bv[0] = (int)b0.x; bv[1] = (int)b0.y; bv[2] = (int)b0.z; bv[3] = (int)b0.w;
                bv[4] = (int)b1.x; bv[5] = (int)b1.y; bv[6] = (int)b1.z; bv[7] = (int)b1.w;
                acc[ni] = __builtin_amdgcn_wmma_i32_16x16x64_iu8(
                    true, av, true, bv, acc[ni], false, false);
            }
        }
    }

    float fs = *finalscale;
    unsigned tcol = (unsigned)t0 + (lane & 15u);
    unsigned cobase = mt * 128u + wv * 16u + (lane >> 4) * 8u;
    #pragma unroll
    for (int ni = 0; ni < 8; ++ni) {
        #pragma unroll
        for (int r = 0; r < 8; ++r) {
            unsigned co = cobase + (unsigned)r;
            unsigned tg = tcol + (unsigned)(ni * 16);
            out[((size_t)b * COUT + co) * T_ + tg] = (float)acc[ni][r] * fs;
        }
    }
}

extern "C" void kernel_launch(void* const* d_in, const int* in_sizes, int n_in,
                              void* d_out, int out_size, void* d_ws, size_t ws_size,
                              hipStream_t stream) {
    (void)n_in; (void)out_size; (void)ws_size;
    const float* x     = (const float*)d_in[0];
    const float* wgt   = (const float*)d_in[1];
    const float* gamma = (const float*)d_in[2];
    float* out = (float*)d_out;
    char* ws = (char*)d_ws;

    unsigned* scalebits = (unsigned*)(ws + WS_SCALEBITS);
    float* wscale       = (float*)(ws + WS_WSCALE);
    float* finalscale   = (float*)(ws + WS_FINAL);
    float* partial      = (float*)(ws + WS_PARTIAL);
    float* inv_rms      = (float*)(ws + WS_INVRMS);
    signed char* xq_t   = (signed char*)(ws + WS_XQ);
    unsigned* wpack     = (unsigned*)(ws + WS_WPACK);
    int nw = in_sizes[1];

    k_init<<<1, 1, 0, stream>>>(scalebits);
    k_rms_max<<<256, 256, 0, stream>>>(x, gamma, inv_rms, scalebits);
    k_wsum<<<256, 256, 0, stream>>>(wgt, partial, nw);
    k_finalize<<<1, 256, 0, stream>>>(partial, scalebits, wscale, finalscale, nw);
    k_actq<<<8192, 256, 0, stream>>>(x, gamma, inv_rms, scalebits, xq_t);
    k_wpack<<<1792, 256, 0, stream>>>(wgt, wscale, wpack);
    k_conv<<<2048, 256, 0, stream>>>(xq_t, wpack, finalscale, out);
}
